// _ProbSparseAttention_6167573037492
// MI455X (gfx1250) — matrix operand order repaired and checked by reading, not compiled, not run on hardware
//
#include <hip/hip_runtime.h>
#include <hip/hip_bf16.h>
#include <cstdint>
#include <cstddef>

typedef __attribute__((ext_vector_type(16))) __bf16 v16bf;
typedef __attribute__((ext_vector_type(8)))  float  v8f;

#define DEV __device__ __forceinline__

constexpr int    Bn  = 4, Ln = 8192, Dm = 1024, Hn = 16, DKn = 64;
constexpr int    Un  = 50, UPn = 64, BHn = Bn * Hn;
constexpr size_t BLn = (size_t)Bn * Ln;           // 32768 rows

// ---------------- workspace layout (bytes) ----------------
constexpr size_t OFF_XB = 0;
constexpr size_t SZ_XB  = BLn * Dm * 2;           // x in bf16
constexpr size_t OFF_WB = OFF_XB + SZ_XB;         // 4 weight matrices bf16
constexpr size_t SZ_W1  = (size_t)Dm * Dm * 2;
constexpr size_t OFF_QB = OFF_WB + 4 * SZ_W1;     // Q bf16 [B][H][L][dk]
constexpr size_t SZ_QKV = BLn * Dm * 2;
constexpr size_t OFF_KB = OFF_QB + SZ_QKV;
constexpr size_t OFF_VB = OFF_KB + SZ_QKV;
constexpr size_t OFF_KS = OFF_VB + SZ_QKV;        // K_samp bf16 [BH][64][dk]
constexpr size_t SZ_KS  = (size_t)BHn * UPn * DKn * 2;
constexpr size_t OFF_M  = OFF_KS + SZ_KS;         // M fp32 [BH][L]
constexpr size_t SZ_M   = (size_t)BHn * Ln * 4;
constexpr size_t OFF_TI = OFF_M + SZ_M;           // top_idx i32 [BH][64]
constexpr size_t SZ_TI  = (size_t)BHn * UPn * 4;
constexpr size_t OFF_QT = OFF_TI + SZ_TI;         // Q_top bf16 [BH][64][dk]
constexpr size_t SZ_QT  = (size_t)BHn * UPn * DKn * 2;
constexpr size_t OFF_VM = OFF_QT + SZ_QT;         // V mean fp32 [BH][dk]
constexpr size_t SZ_VM  = (size_t)BHn * DKn * 4;
constexpr size_t OFF_CT = OFF_VM + SZ_VM;         // ctx_top fp32 [BH][64][dk]
constexpr size_t SZ_CT  = (size_t)BHn * UPn * DKn * 4;
constexpr size_t OFF_CX = OFF_CT + SZ_CT;         // ctx bf16 [B][L][D]

// ---------------- fragment helpers ----------------
DEV v16bf frag_from2(const __bf16* p0, const __bf16* p1) {
  union { uint4 u[2]; v16bf v; } u;
  u.u[0] = *(const uint4*)p0;
  u.u[1] = *(const uint4*)p1;
  return u.v;
}
// A 16x32 bf16: lane-half h holds K {16h..16h+15} contiguous
DEV v16bf afrag(const __bf16* row, int half) {
  return frag_from2(row + half * 16, row + half * 16 + 8);
}
// B 32x16 bf16: lane-half h holds K {16h..16h+15} contiguous
DEV v16bf bfrag(const __bf16* col, int off) {
  return frag_from2(col + off, col + off + 8);
}
DEV v8f wmma_bf16(v16bf a, v16bf b, v8f c) {
  return __builtin_amdgcn_wmma_f32_16x16x32_bf16(false, a, false, b,
                                                 (short)0, c, false, false);
}

// ---------------- CDNA5 async memory->LDS copy (ASYNCcnt) ----------------
DEV void g2l_b128(const void* lds_dst, const void* gsrc) {
  unsigned int       loff = (unsigned int)(uintptr_t)lds_dst;   // LDS = addr[31:0]
  unsigned long long ga   = (unsigned long long)(uintptr_t)gsrc;
  asm volatile("global_load_async_to_lds_b128 %0, %1, off"
               :: "v"(loff), "v"(ga) : "memory");
}
DEV void wait_async0() {
  asm volatile("s_wait_asynccnt 0x0" ::: "memory");
}

// ---------------- fp32 -> bf16 conversion ----------------
__global__ __launch_bounds__(256) void cvt_bf16_k(const float* __restrict__ src,
                                                  __bf16* __restrict__ dst, int n) {
  int i = blockIdx.x * 256 + threadIdx.x;
  if (i < n) dst[i] = (__bf16)src[i];
}

// ---------------- tiled bf16 GEMM: C = A * W^T + bias ----------------
// A [32768][1024] bf16 row-major, W [1024][1024] bf16 row-major (out,K)
// 128x64 tile, 128 threads (4 waves), double-buffered async LDS staging.
// mode 0: write bf16 head-major [B][H][L][dk]; mode 1: write fp32 [M][1024]
__global__ __launch_bounds__(128) void gemm_bf16_k(
    const __bf16* __restrict__ A, const __bf16* __restrict__ Wt,
    const float* __restrict__ bias, __bf16* __restrict__ outb,
    float* __restrict__ outf, int mode) {
  __shared__ __bf16 As[2][128][40];
  __shared__ __bf16 Bs[2][64][40];
  const int tid  = threadIdx.x;
  const int wave = tid >> 5, lane = tid & 31;
  const int half = lane >> 4, ln = lane & 15;
  const size_t m0 = (size_t)blockIdx.x * 128;
  const int    n0 = blockIdx.y * 64;
  v8f acc[2][4] = {{v8f{}, v8f{}, v8f{}, v8f{}}, {v8f{}, v8f{}, v8f{}, v8f{}}};

  auto issue_tiles = [&](int b, int k0) {
#pragma unroll
    for (int i = 0; i < 4; ++i) {              // A: 128x32 bf16
      int v = tid + i * 128;
      int r = v >> 2, c = (v & 3) * 8;
      g2l_b128(&As[b][r][c], A + (m0 + r) * Dm + k0 + c);
    }
#pragma unroll
    for (int i = 0; i < 2; ++i) {              // B: 64x32 bf16
      int v = tid + i * 128;
      int r = v >> 2, c = (v & 3) * 8;
      g2l_b128(&Bs[b][r][c], Wt + (size_t)(n0 + r) * Dm + k0 + c);
    }
  };

  issue_tiles(0, 0);
  for (int k0 = 0; k0 < Dm; k0 += 32) {
    const int buf = (k0 >> 5) & 1;
    wait_async0();           // my copies into `buf` have landed
    __syncthreads();         // everyone's landed; everyone done reading buf^1
    if (k0 + 32 < Dm) issue_tiles(buf ^ 1, k0 + 32);
    v16bf a0 = afrag(&As[buf][wave * 32 + ln][0], half);
    v16bf a1 = afrag(&As[buf][wave * 32 + 16 + ln][0], half);
#pragma unroll
    for (int nt = 0; nt < 4; ++nt) {
      v16bf b = bfrag(&Bs[buf][nt * 16 + ln][0], half * 16);
      acc[0][nt] = wmma_bf16(a0, b, acc[0][nt]);
      acc[1][nt] = wmma_bf16(a1, b, acc[1][nt]);
    }
  }

#pragma unroll
  for (int ms = 0; ms < 2; ++ms) {
#pragma unroll
    for (int nt = 0; nt < 4; ++nt) {
      int gn = n0 + nt * 16 + ln;
      float bb = bias[gn];
#pragma unroll
      for (int i = 0; i < 8; ++i) {
        size_t gm = m0 + wave * 32 + ms * 16 + half * 8 + i;
        float v = acc[ms][nt][i] + bb;
        if (mode == 0) {
          int b = (int)(gm >> 13), l = (int)(gm & 8191);
          int h = gn >> 6, d = gn & 63;
          outb[(((size_t)(b * Hn + h)) * Ln + l) * DKn + d] = (__bf16)v;
        } else {
          outf[gm * Dm + gn] = v;
        }
      }
    }
  }
}

// ---------------- gather K_samp (pad rows 50..63 with 0) ----------------
__global__ __launch_bounds__(256) void ksamp_gather_k(
    const __bf16* __restrict__ Kb, const int* __restrict__ idx_k,
    __bf16* __restrict__ Ksamp) {
  int bh = blockIdx.x;
  for (int e = threadIdx.x; e < UPn * DKn; e += 256) {
    int j = e >> 6, d = e & 63;
    __bf16 v = (__bf16)0.0f;
    if (j < Un) {
      int l = idx_k[j];
      v = Kb[((size_t)bh * Ln + l) * DKn + d];
    }
    Ksamp[(size_t)bh * (UPn * DKn) + e] = v;
  }
}

// ---------------- M = scale*(max - mean) over 50 sampled keys ----------------
__global__ __launch_bounds__(256) void scores_m_k(
    const __bf16* __restrict__ Qb, const __bf16* __restrict__ Ksamp,
    float* __restrict__ Mout) {
  __shared__ float ks[Un][DKn + 1];
  int bh = blockIdx.x;
  for (int e = threadIdx.x; e < Un * DKn; e += 256) {
    int j = e >> 6, d = e & 63;
    ks[j][d] = (float)Ksamp[((size_t)bh * UPn + j) * DKn + d];
  }
  __syncthreads();
  int l = blockIdx.y * 256 + threadIdx.x;
  const __bf16* q = Qb + ((size_t)bh * Ln + l) * DKn;
  float qr[DKn];
#pragma unroll
  for (int d = 0; d < DKn; ++d) qr[d] = (float)q[d];
  float mx = -3.0e38f, sm = 0.f;
  for (int j = 0; j < Un; ++j) {
    float s = 0.f;
#pragma unroll
    for (int d = 0; d < DKn; ++d) s = fmaf(qr[d], ks[j][d], s);
    mx = fmaxf(mx, s);
    sm += s;
  }
  Mout[(size_t)bh * Ln + l] = 0.125f * (mx - sm * (1.0f / 50.0f));
}

// ---------------- top-50 selection + Q_top gather ----------------
__global__ __launch_bounds__(256) void topk_qtop_k(
    const float* __restrict__ Mv, const __bf16* __restrict__ Qb,
    int* __restrict__ topidx, __bf16* __restrict__ Qtop) {
  __shared__ float vals[Ln];
  __shared__ float rv[256];
  __shared__ int   ri[256];
  __shared__ int   sel[UPn];
  int bh = blockIdx.x, tid = threadIdx.x;
  for (int i = tid; i < Ln; i += 256) vals[i] = Mv[(size_t)bh * Ln + i];
  __syncthreads();
  for (int t = 0; t < Un; ++t) {
    float bm = -3.0e38f; int bi = 0;
    for (int i = tid; i < Ln; i += 256) {
      float v = vals[i];
      if (v > bm) { bm = v; bi = i; }
    }
    rv[tid] = bm; ri[tid] = bi;
    __syncthreads();
    for (int s = 128; s > 0; s >>= 1) {
      if (tid < s && rv[tid + s] > rv[tid]) { rv[tid] = rv[tid + s]; ri[tid] = ri[tid + s]; }
      __syncthreads();
    }
    if (tid == 0) { sel[t] = ri[0]; vals[ri[0]] = -3.0e38f; }
    __syncthreads();
  }
  if (tid < UPn - Un) sel[Un + tid] = 0;
  __syncthreads();
  if (tid < UPn) topidx[bh * UPn + tid] = sel[tid];
  for (int e = tid; e < UPn * DKn; e += 256) {
    int j = e >> 6, d = e & 63;
    __bf16 v = (__bf16)0.0f;
    if (j < Un) v = Qb[((size_t)bh * Ln + sel[j]) * DKn + d];
    Qtop[(size_t)bh * (UPn * DKn) + e] = v;
  }
}

// ---------------- V mean over L ----------------
__global__ __launch_bounds__(256) void vmean_k(const __bf16* __restrict__ Vb,
                                               float* __restrict__ Vmean) {
  __shared__ float part[256];
  int bh = blockIdx.x, tid = threadIdx.x;
  int d = tid & 63, p = tid >> 6;
  float s = 0.f;
  for (int l = p * (Ln / 4); l < (p + 1) * (Ln / 4); ++l)
    s += (float)Vb[((size_t)bh * Ln + l) * DKn + d];
  part[tid] = s;
  __syncthreads();
  if (tid < DKn)
    Vmean[bh * DKn + tid] =
        (part[tid] + part[tid + 64] + part[tid + 128] + part[tid + 192]) *
        (1.0f / (float)Ln);
}

// ---------------- flash attention for the 64 (50) top queries ----------------
DEV void compute_S(const __bf16 (*Kc)[72], int ln, int half, v16bf aq0,
                   v16bf aq1, v8f* s) {
#pragma unroll
  for (int nt = 0; nt < 8; ++nt) {
    const __bf16* kp = &Kc[nt * 16 + ln][0];
    v16bf b0 = bfrag(kp, half * 16);        // K(d) 0..31
    v16bf b1 = bfrag(kp, 32 + half * 16);   // K(d) 32..63
    v8f t = {};
    t = wmma_bf16(aq0, b0, t);
    t = wmma_bf16(aq1, b1, t);
    s[nt] = t;
  }
}

__global__ __launch_bounds__(128) void attn_topq_k(
    const __bf16* __restrict__ Qtop, const __bf16* __restrict__ Kb,
    const __bf16* __restrict__ Vb, float* __restrict__ ctxtop) {
  __shared__ __bf16 Kc[128][72];     // K chunk, row-major (l, d)
  __shared__ __bf16 Vt[64][136];     // V chunk transposed (d, l)
  __shared__ __bf16 Pl[4][16][136];  // per-wave P staging (C->A layout)
  const int bh = blockIdx.x;
  const int tid = threadIdx.x;
  const int wave = tid >> 5, lane = tid & 31;
  const int half = lane >> 4, ln = lane & 15;
  const float scale = 0.125f;
  const __bf16* qr = Qtop + ((size_t)bh * UPn + wave * 16 + ln) * DKn;
  v16bf aq0 = afrag(qr, half);
  v16bf aq1 = afrag(qr + 32, half);
  float rmax[8], rl[8];
#pragma unroll
  for (int i = 0; i < 8; ++i) { rmax[i] = -3.0e38f; rl[i] = 0.f; }
  const __bf16* Kbh = Kb + (size_t)bh * Ln * DKn;
  const __bf16* Vbh = Vb + (size_t)bh * Ln * DKn;

  // pass 1: row max + sum of exp (online)
  for (int l0 = 0; l0 < Ln; l0 += 128) {
    __syncthreads();
#pragma unroll
    for (int i = 0; i < 8; ++i) {
      int v = tid + i * 128;
      int r = v >> 3, c = (v & 7) * 8;
      g2l_b128(&Kc[r][c], Kbh + (size_t)(l0 + r) * DKn + c);
    }
    wait_async0();
    __syncthreads();
    v8f s[8];
    compute_S(Kc, ln, half, aq0, aq1, s);
#pragma unroll
    for (int i = 0; i < 8; ++i) {
      float m = s[0][i];
#pragma unroll
      for (int nt = 1; nt < 8; ++nt) m = fmaxf(m, s[nt][i]);
      m *= scale;
#pragma unroll
      for (int sh = 1; sh < 16; sh <<= 1) m = fmaxf(m, __shfl_xor(m, sh, 32));
      float mn = fmaxf(rmax[i], m);
      float e = 0.f;
#pragma unroll
      for (int nt = 0; nt < 8; ++nt) e += __expf(s[nt][i] * scale - mn);
#pragma unroll
      for (int sh = 1; sh < 16; sh <<= 1) e += __shfl_xor(e, sh, 32);
      rl[i] = rl[i] * __expf(rmax[i] - mn) + e;
      rmax[i] = mn;
    }
  }
  float invl[8];
#pragma unroll
  for (int i = 0; i < 8; ++i) invl[i] = 1.0f / rl[i];

  // pass 2: P = exp(S-m)/l  (C-layout -> LDS -> A-layout), ctx += P*V
  v8f cacc[4] = {v8f{}, v8f{}, v8f{}, v8f{}};
  for (int l0 = 0; l0 < Ln; l0 += 128) {
    __syncthreads();
#pragma unroll
    for (int i = 0; i < 8; ++i) {
      int v = tid + i * 128;
      int r = v >> 3, c = (v & 7) * 8;
      g2l_b128(&Kc[r][c], Kbh + (size_t)(l0 + r) * DKn + c);
      // V chunk: transpose into Vt while staging
      uint4 raw = *(const uint4*)(Vbh + (size_t)(l0 + r) * DKn + c);
      const __bf16* ve = (const __bf16*)&raw;
#pragma unroll
      for (int e2 = 0; e2 < 8; ++e2) Vt[c + e2][r] = ve[e2];
    }
    wait_async0();
    __syncthreads();
    v8f s[8];
    compute_S(Kc, ln, half, aq0, aq1, s);
#pragma unroll
    for (int nt = 0; nt < 8; ++nt) {
#pragma unroll
      for (int i = 0; i < 8; ++i) {
        float p = __expf(s[nt][i] * scale - rmax[i]) * invl[i];
        Pl[wave][half * 8 + i][nt * 16 + ln] = (__bf16)p;
      }
    }
    __syncthreads();
#pragma unroll
    for (int dt = 0; dt < 4; ++dt) {
      v8f c = cacc[dt];
#pragma unroll
      for (int kk = 0; kk < 4; ++kk) {
        v16bf ap = afrag(&Pl[wave][ln][kk * 32], half);
        v16bf bv = bfrag(&Vt[dt * 16 + ln][0], kk * 32 + half * 16);
        c = wmma_bf16(ap, bv, c);
      }
      cacc[dt] = c;
    }
  }
#pragma unroll
  for (int dt = 0; dt < 4; ++dt) {
#pragma unroll
    for (int i = 0; i < 8; ++i) {
      int row = wave * 16 + half * 8 + i;
      ctxtop[((size_t)bh * UPn + row) * DKn + dt * 16 + ln] = cacc[dt][i];
    }
  }
}

// ---------------- ctx = broadcast(V mean), token-major bf16 ----------------
__global__ __launch_bounds__(256) void ctx_fill_k(const float* __restrict__ Vmean,
                                                  __bf16* __restrict__ ctxb) {
  size_t idx = (size_t)blockIdx.x * 256 + threadIdx.x;
  if (idx >= BLn * Dm) return;
  int n = (int)(idx & 1023);
  size_t bl = idx >> 10;
  int b = (int)(bl >> 13);
  int h = n >> 6, d = n & 63;
  ctxb[idx] = (__bf16)Vmean[(b * Hn + h) * DKn + d];
}

// ---------------- scatter ctx_top into ctx ----------------
__global__ __launch_bounds__(256) void ctx_scatter_k(
    const int* __restrict__ topidx, const float* __restrict__ ctxtop,
    __bf16* __restrict__ ctxb) {
  int bh = blockIdx.x;
  int b = bh >> 4, h = bh & 15;
  for (int e = threadIdx.x; e < Un * DKn; e += 256) {
    int j = e >> 6, d = e & 63;
    int l = topidx[bh * UPn + j];
    ctxb[((size_t)(b * Ln + l)) * Dm + h * DKn + d] =
        (__bf16)ctxtop[((size_t)bh * UPn + j) * DKn + d];
  }
}

// ---------------- host launch ----------------
extern "C" void kernel_launch(void* const* d_in, const int* in_sizes, int n_in,
                              void* d_out, int out_size, void* d_ws,
                              size_t ws_size, hipStream_t stream) {
  (void)in_sizes; (void)n_in; (void)out_size; (void)ws_size;
  const float* x   = (const float*)d_in[0];
  const float* Wq  = (const float*)d_in[1];
  const float* bq  = (const float*)d_in[2];
  const float* Wk  = (const float*)d_in[3];
  const float* bk  = (const float*)d_in[4];
  const float* Wv  = (const float*)d_in[5];
  const float* bv  = (const float*)d_in[6];
  const float* Wo  = (const float*)d_in[7];
  const float* bo  = (const float*)d_in[8];
  const int* idx_k = (const int*)d_in[9];
  float* out = (float*)d_out;
  char* ws = (char*)d_ws;

  __bf16* xb    = (__bf16*)(ws + OFF_XB);
  __bf16* wb0   = (__bf16*)(ws + OFF_WB);
  __bf16* wb1   = (__bf16*)(ws + OFF_WB + SZ_W1);
  __bf16* wb2   = (__bf16*)(ws + OFF_WB + 2 * SZ_W1);
  __bf16* wb3   = (__bf16*)(ws + OFF_WB + 3 * SZ_W1);
  __bf16* Qb    = (__bf16*)(ws + OFF_QB);
  __bf16* Kb    = (__bf16*)(ws + OFF_KB);
  __bf16* Vb    = (__bf16*)(ws + OFF_VB);
  __bf16* Ksamp = (__bf16*)(ws + OFF_KS);
  float*  Mw    = (float*)(ws + OFF_M);
  int*    ti    = (int*)(ws + OFF_TI);
  __bf16* Qt    = (__bf16*)(ws + OFF_QT);
  float*  Vm    = (float*)(ws + OFF_VM);
  float*  Ct    = (float*)(ws + OFF_CT);
  __bf16* Cx    = (__bf16*)(ws + OFF_CX);

  const int nX = (int)(BLn * Dm);      // 33554432
  const int nW = Dm * Dm;              // 1048576
  cvt_bf16_k<<<(nX + 255) / 256, 256, 0, stream>>>(x, xb, nX);
  cvt_bf16_k<<<(nW + 255) / 256, 256, 0, stream>>>(Wq, wb0, nW);
  cvt_bf16_k<<<(nW + 255) / 256, 256, 0, stream>>>(Wk, wb1, nW);
  cvt_bf16_k<<<(nW + 255) / 256, 256, 0, stream>>>(Wv, wb2, nW);
  cvt_bf16_k<<<(nW + 255) / 256, 256, 0, stream>>>(Wo, wb3, nW);

  dim3 gg((unsigned)(BLn / 128), Dm / 64);
  gemm_bf16_k<<<gg, 128, 0, stream>>>(xb, wb0, bq, Qb, nullptr, 0);
  gemm_bf16_k<<<gg, 128, 0, stream>>>(xb, wb1, bk, Kb, nullptr, 0);
  gemm_bf16_k<<<gg, 128, 0, stream>>>(xb, wb2, bv, Vb, nullptr, 0);

  ksamp_gather_k<<<BHn, 256, 0, stream>>>(Kb, idx_k, Ksamp);
  scores_m_k<<<dim3(BHn, Ln / 256), 256, 0, stream>>>(Qb, Ksamp, Mw);
  topk_qtop_k<<<BHn, 256, 0, stream>>>(Mw, Qb, ti, Qt);
  vmean_k<<<BHn, 256, 0, stream>>>(Vb, Vm);
  attn_topq_k<<<BHn, 128, 0, stream>>>(Qt, Kb, Vb, Ct);

  ctx_fill_k<<<(nX + 255) / 256, 256, 0, stream>>>(Vm, Cx);
  ctx_scatter_k<<<BHn, 256, 0, stream>>>(ti, Ct, Cx);

  gemm_bf16_k<<<gg, 128, 0, stream>>>(Cx, wb3, bo, nullptr, out, 1);
}
